// Attention_51874615001506
// MI455X (gfx1250) — matrix operand order repaired and checked by reading, not compiled, not run on hardware
//
#include <hip/hip_runtime.h>

#define DIM     1024
#define NHEADS  16
#define HDIM    64
#define SEQ     2048
#define BATCH   2
#define MTOT    (BATCH * SEQ)

typedef __bf16 bf16_t;
typedef bf16_t v16bf __attribute__((ext_vector_type(16)));
typedef float  v8f   __attribute__((ext_vector_type(8)));
typedef int    v4i_t __attribute__((__vector_size__(16)));   // matches builtin param

#define AS1 __attribute__((address_space(1)))
#define AS3 __attribute__((address_space(3)))

#if defined(__gfx1250__) && __has_builtin(__builtin_amdgcn_global_load_async_to_lds_b128) && \
    __has_builtin(__builtin_amdgcn_s_wait_asynccnt)
#define USE_ASYNC_LDS 1
#else
#define USE_ASYNC_LDS 0
#endif

union Frag  { v16bf v; unsigned int u[8]; };
union CFrag { v8f   v; float        f[8]; };

__device__ __forceinline__ v8f wmma_bf16(const Frag& a, const Frag& b, v8f c) {
    // 8 args: (neg_a, A, neg_b, B, c_mod, C, reuse_a, reuse_b)
    return __builtin_amdgcn_wmma_f32_16x16x32_bf16(false, a.v, false, b.v,
                                                   (short)0, c, false, false);
}

#if USE_ASYNC_LDS
__device__ __forceinline__ void async_copy_b128(const void* gsrc, void* ldst) {
    __builtin_amdgcn_global_load_async_to_lds_b128(
        (AS1 v4i_t*)gsrc, (AS3 v4i_t*)ldst, 0, 0);
}
#endif

// ---------------------------------------------------------------------------
// Kernel 1: fused QKV projection (q|k|v = x @ W^T) + RoPE on q,k.
// grid = (DIM/64, MTOT/128, 3), block = 128 (4 waves).
// Block tile 128(M) x 64(N); each wave owns 32 rows (two 16-row strips) so every
// B fragment is reused twice: 8 wmma per k-step vs 6 fragment loads.
// Output layout: (B, H, S, Dh) bf16.
// ---------------------------------------------------------------------------
__global__ __launch_bounds__(128)
void qkv_rope_gemm(const float* __restrict__ x,
                   const float* __restrict__ Wq,
                   const float* __restrict__ Wk,
                   const float* __restrict__ Wv,
                   bf16_t* __restrict__ qws,
                   bf16_t* __restrict__ kws,
                   bf16_t* __restrict__ vws)
{
    __shared__ __align__(16) bf16_t As[128][32];
    __shared__ __align__(16) bf16_t Bs[64][32];

    const int tid  = threadIdx.x;
    const int lane = tid & 31;
    const int wave = tid >> 5;
    const int n0   = blockIdx.x * 64;
    const int m0   = blockIdx.y * 128;
    const int z    = blockIdx.z;

    const float* W   = (z == 0) ? Wq : (z == 1) ? Wk : Wv;
    bf16_t*      ows = (z == 0) ? qws : (z == 1) ? kws : vws;

    CFrag acc[2][4];
    #pragma unroll
    for (int s = 0; s < 2; ++s)
        #pragma unroll
        for (int t = 0; t < 4; ++t)
            #pragma unroll
            for (int i = 0; i < 8; ++i) acc[s][t].f[i] = 0.0f;

    for (int k0 = 0; k0 < DIM; k0 += 32) {
        __syncthreads();
        // A tile 128x32: vector f32 loads + native bf16 convert (v_cvt_pk_bf16_f32).
        for (int i = tid; i < 128 * 16; i += 128) {
            int r = i >> 4, c = (i & 15) * 2;
            float2 f = *(const float2*)&x[(size_t)(m0 + r) * DIM + k0 + c];
            As[r][c]     = (bf16_t)f.x;
            As[r][c + 1] = (bf16_t)f.y;
        }
        // B tile 64x32 from weights.
        for (int i = tid; i < 64 * 16; i += 128) {
            int r = i >> 4, c = (i & 15) * 2;
            float2 f = *(const float2*)&W[(size_t)(n0 + r) * DIM + k0 + c];
            Bs[r][c]     = (bf16_t)f.x;
            Bs[r][c + 1] = (bf16_t)f.y;
        }
        __syncthreads();

        // A fragments (16x32 bf16 ISA layout): lane = row, K {0-7,16-23}/{8-15,24-31}.
        Frag a[2];
        #pragma unroll
        for (int s = 0; s < 2; ++s) {
            const unsigned int* ap =
                (const unsigned int*)&As[wave * 32 + s * 16 + (lane & 15)][0];
            #pragma unroll
            for (int v = 0; v < 8; ++v)
                a[s].u[v] = ap[(v & 3) + ((v >= 4) ? 8 : 0) + ((lane & 16) ? 4 : 0)];
        }
        #pragma unroll
        for (int t = 0; t < 4; ++t) {
            // B fragment (32x16 bf16): lane = col, K {0-7,16-23}/{8-15,24-31}.
            Frag b;
            const unsigned int* bp = (const unsigned int*)&Bs[t * 16 + (lane & 15)][0];
            #pragma unroll
            for (int v = 0; v < 8; ++v)
                b.u[v] = bp[(v & 3) + ((v >= 4) ? 8 : 0) + ((lane & 16) ? 4 : 0)];
            acc[0][t].v = wmma_bf16(a[0], b, acc[0][t].v);
            acc[1][t].v = wmma_bf16(a[1], b, acc[1][t].v);
        }
    }

    // RoPE (q,k only) directly on WMMA C layout, then store bf16 (B,H,S,Dh).
    const bool doRope = (z < 2);
    #pragma unroll
    for (int s = 0; s < 2; ++s) {
        #pragma unroll
        for (int t = 0; t < 4; ++t) {
            #pragma unroll
            for (int r = 0; r < 8; ++r) {
                int m = m0 + wave * 32 + s * 16 + r + ((lane & 16) ? 8 : 0);
                int n = n0 + t * 16 + (lane & 15);
                float val = acc[s][t].f[r];
                if (doRope) {
                    float partner = __shfl_xor(val, 1, 32); // col^1 == lane^1 in C layout
                    int   sq  = m & (SEQ - 1);
                    int   j   = (n & 63) >> 1;
                    float inv = __expf(-(float)j * 0.28782313662425574f); // theta^(-2j/64)
                    float ang = (float)sq * inv;
                    float cs  = __cosf(ang), sn = __sinf(ang);
                    val = (n & 1) ? (partner * sn + val * cs)
                                  : (val * cs - partner * sn);
                }
                int b = m >> 11, s2 = m & (SEQ - 1), h = n >> 6, d = n & 63;
                ows[((size_t)(b * NHEADS + h) * SEQ + s2) * HDIM + d] = (bf16_t)val;
            }
        }
    }
}

// ---------------------------------------------------------------------------
// Kernel 2: streaming causal attention (flash style), fp32 softmax state.
// grid = (SEQ/64, NHEADS, BATCH), block = 128 (4 waves; wave = 16 query rows).
// Scores never touch global memory. K tile staged with async load-to-LDS.
// ---------------------------------------------------------------------------
__global__ __launch_bounds__(128)
void attention_kernel(const bf16_t* __restrict__ qws,
                      const bf16_t* __restrict__ kws,
                      const bf16_t* __restrict__ vws,
                      bf16_t* __restrict__ aws)
{
    __shared__ __align__(16) bf16_t Ks[32][64];     // keys row-major
    __shared__ __align__(16) bf16_t Vt[64][32];     // V transposed: [dim][key]
    __shared__ __align__(16) bf16_t Ps[4][16][32];  // per-wave P transpose scratch

    const int tid   = threadIdx.x;
    const int lane  = tid & 31;
    const int wave  = tid >> 5;
    const int qbase = blockIdx.x * 64;
    const int h     = blockIdx.y;
    const int b     = blockIdx.z;
    const size_t headOff = (size_t)(b * NHEADS + h) * SEQ * HDIM;

    // Q A-fragments for this wave's 16 rows (Dh=64 -> two K=32 fragments), in VGPRs.
    Frag aq[2];
    {
        const unsigned int* qp = (const unsigned int*)
            (qws + headOff + (size_t)(qbase + wave * 16 + (lane & 15)) * HDIM);
        #pragma unroll
        for (int j = 0; j < 2; ++j)
            #pragma unroll
            for (int v = 0; v < 8; ++v)
                aq[j].u[v] = qp[j * 16 + (v & 3) + ((v >= 4) ? 8 : 0) + ((lane & 16) ? 4 : 0)];
    }

    float mrow[8], lrow[8];
    CFrag o[4];
    #pragma unroll
    for (int r = 0; r < 8; ++r) { mrow[r] = -3e38f; lrow[r] = 0.0f; }
    #pragma unroll
    for (int t = 0; t < 4; ++t)
        #pragma unroll
        for (int i = 0; i < 8; ++i) o[t].f[i] = 0.0f;

    const int qmax = qbase + 63;
    for (int kb = 0; kb <= qmax; kb += 32) {
        __syncthreads();
        // Stage K tile (32 keys x 64 dims): contiguous 4 KB -> async copy to LDS.
        {
            const bf16_t* src = kws + headOff + (size_t)kb * HDIM;
#if USE_ASYNC_LDS
            for (int i = tid; i < 256; i += 128)
                async_copy_b128(src + i * 8, &Ks[0][0] + i * 8);
#else
            const uint4* s4 = (const uint4*)src;
            uint4*       d4 = (uint4*)&Ks[0][0];
            for (int i = tid; i < 256; i += 128) d4[i] = s4[i];
#endif
        }
        // Stage V transposed so P*V B-fragments are contiguous LDS reads.
        {
            const bf16_t* src = vws + headOff + (size_t)kb * HDIM;
            for (int i = tid; i < 32 * 64; i += 128)
                Vt[i & 63][i >> 6] = src[i];
        }
        if (kb + 32 <= qmax) {  // hint next tiles into cache (global_prefetch_b8)
            __builtin_prefetch(kws + headOff + (size_t)(kb + 32) * HDIM, 0, 1);
            __builtin_prefetch(vws + headOff + (size_t)(kb + 32) * HDIM, 0, 1);
        }
#if USE_ASYNC_LDS
        __builtin_amdgcn_s_wait_asynccnt(0);
#endif
        __syncthreads();

        // S = Q K^T : two 16-key n-tiles, each K-dim 64 = 2 x wmma(K=32).
        CFrag sc[2];
        #pragma unroll
        for (int nt = 0; nt < 2; ++nt) {
            #pragma unroll
            for (int i = 0; i < 8; ++i) sc[nt].f[i] = 0.0f;
            const unsigned int* kp = (const unsigned int*)&Ks[nt * 16 + (lane & 15)][0];
            #pragma unroll
            for (int j = 0; j < 2; ++j) {
                Frag bk;
                #pragma unroll
                for (int v = 0; v < 8; ++v)
                    bk.u[v] = kp[j * 16 + (v & 3) + ((v >= 4) ? 8 : 0) + ((lane & 16) ? 4 : 0)];
                sc[nt].v = wmma_bf16(aq[j], bk, sc[nt].v);
            }
        }

        // Causal mask + online softmax (fp32); P staged per-wave in LDS for transpose.
        #pragma unroll
        for (int r = 0; r < 8; ++r) {
            int qrow = qbase + wave * 16 + r + ((lane & 16) ? 8 : 0);
            int key0 = kb + (lane & 15);
            float s0 = (key0      <= qrow) ? sc[0].f[r] * 0.125f : -3e30f;
            float s1 = (key0 + 16 <= qrow) ? sc[1].f[r] * 0.125f : -3e30f;
            float tmax = fmaxf(s0, s1);
            #pragma unroll
            for (int off = 1; off < 16; off <<= 1)
                tmax = fmaxf(tmax, __shfl_xor(tmax, off, 32));
            float mnew  = fmaxf(mrow[r], tmax);
            float alpha = __expf(mrow[r] - mnew);
            float p0 = __expf(s0 - mnew);
            float p1 = __expf(s1 - mnew);
            float psum = p0 + p1;
            #pragma unroll
            for (int off = 1; off < 16; off <<= 1)
                psum += __shfl_xor(psum, off, 32);
            lrow[r] = lrow[r] * alpha + psum;
            mrow[r] = mnew;
            #pragma unroll
            for (int t = 0; t < 4; ++t) o[t].f[r] *= alpha;
            int prow = r + ((lane & 16) ? 8 : 0);
            Ps[wave][prow][lane & 15]        = (bf16_t)p0;
            Ps[wave][prow][16 + (lane & 15)] = (bf16_t)p1;
        }

        // O += P (16x32) * V (32x64) : 4 wmma. P re-read in A layout from LDS.
        Frag ap;
        {
            const unsigned int* pp = (const unsigned int*)&Ps[wave][lane & 15][0];
            #pragma unroll
            for (int v = 0; v < 8; ++v)
                ap.u[v] = pp[(v & 3) + ((v >= 4) ? 8 : 0) + ((lane & 16) ? 4 : 0)];
        }
        #pragma unroll
        for (int t = 0; t < 4; ++t) {
            Frag bv;
            const unsigned int* vp = (const unsigned int*)&Vt[t * 16 + (lane & 15)][0];
            #pragma unroll
            for (int v = 0; v < 8; ++v)
                bv.u[v] = vp[(v & 3) + ((v >= 4) ? 8 : 0) + ((lane & 16) ? 4 : 0)];
            o[t].v = wmma_bf16(ap, bv, o[t].v);
        }
    }

    // Normalize and store bf16 in (B, S, D) for the output projection.
    #pragma unroll
    for (int t = 0; t < 4; ++t) {
        #pragma unroll
        for (int r = 0; r < 8; ++r) {
            int qrow = qbase + wave * 16 + r + ((lane & 16) ? 8 : 0);
            int col  = h * HDIM + t * 16 + (lane & 15);
            aws[(size_t)(b * SEQ + qrow) * DIM + col] = (bf16_t)(o[t].f[r] / lrow[r]);
        }
    }
}

// ---------------------------------------------------------------------------
// Kernel 3: y = attn_out(bf16) @ Wo^T -> fp32 d_out.
// Same 128x64 tiling as kernel 1; bf16 A tile staged via async load-to-LDS.
// ---------------------------------------------------------------------------
__global__ __launch_bounds__(128)
void oproj_gemm(const bf16_t* __restrict__ a,
                const float* __restrict__ Wo,
                float* __restrict__ out)
{
    __shared__ __align__(16) bf16_t As[128][32];
    __shared__ __align__(16) bf16_t Bs[64][32];

    const int tid  = threadIdx.x;
    const int lane = tid & 31;
    const int wave = tid >> 5;
    const int n0   = blockIdx.x * 64;
    const int m0   = blockIdx.y * 128;

    CFrag acc[2][4];
    #pragma unroll
    for (int s = 0; s < 2; ++s)
        #pragma unroll
        for (int t = 0; t < 4; ++t)
            #pragma unroll
            for (int i = 0; i < 8; ++i) acc[s][t].f[i] = 0.0f;

    for (int k0 = 0; k0 < DIM; k0 += 32) {
        __syncthreads();
        // A tile 128x32 bf16: each row is a 64 B contiguous run -> 4 b128 async copies.
        {
#if USE_ASYNC_LDS
            for (int i = tid; i < 512; i += 128) {
                int r = i >> 2, q = i & 3;
                async_copy_b128(a + (size_t)(m0 + r) * DIM + k0 + q * 8, &As[r][q * 8]);
            }
#else
            for (int i = tid; i < 512; i += 128) {
                int r = i >> 2, q = i & 3;
                *(uint4*)&As[r][q * 8] =
                    *(const uint4*)(a + (size_t)(m0 + r) * DIM + k0 + q * 8);
            }
#endif
        }
        for (int i = tid; i < 64 * 16; i += 128) {
            int r = i >> 4, c = (i & 15) * 2;
            float2 f = *(const float2*)&Wo[(size_t)(n0 + r) * DIM + k0 + c];
            Bs[r][c]     = (bf16_t)f.x;
            Bs[r][c + 1] = (bf16_t)f.y;
        }
#if USE_ASYNC_LDS
        __builtin_amdgcn_s_wait_asynccnt(0);
#endif
        __syncthreads();

        Frag af[2];
        #pragma unroll
        for (int s = 0; s < 2; ++s) {
            const unsigned int* ap =
                (const unsigned int*)&As[wave * 32 + s * 16 + (lane & 15)][0];
            #pragma unroll
            for (int v = 0; v < 8; ++v)
                af[s].u[v] = ap[(v & 3) + ((v >= 4) ? 8 : 0) + ((lane & 16) ? 4 : 0)];
        }
        #pragma unroll
        for (int t = 0; t < 4; ++t) {
            Frag bfr;
            const unsigned int* bp = (const unsigned int*)&Bs[t * 16 + (lane & 15)][0];
            #pragma unroll
            for (int v = 0; v < 8; ++v)
                bfr.u[v] = bp[(v & 3) + ((v >= 4) ? 8 : 0) + ((lane & 16) ? 4 : 0)];
            acc[0][t].v = wmma_bf16(af[0], bfr, acc[0][t].v);
            acc[1][t].v = wmma_bf16(af[1], bfr, acc[1][t].v);
        }
    }

    #pragma unroll
    for (int s = 0; s < 2; ++s) {
        #pragma unroll
        for (int t = 0; t < 4; ++t) {
            #pragma unroll
            for (int r = 0; r < 8; ++r) {
                int m = m0 + wave * 32 + s * 16 + r + ((lane & 16) ? 8 : 0);
                int n = n0 + t * 16 + (lane & 15);
                out[(size_t)m * DIM + n] = acc[s][t].f[r];
            }
        }
    }
}

// ---------------------------------------------------------------------------
extern "C" void kernel_launch(void* const* d_in, const int* in_sizes, int n_in,
                              void* d_out, int out_size, void* d_ws, size_t ws_size,
                              hipStream_t stream)
{
    (void)in_sizes; (void)n_in; (void)out_size; (void)ws_size;
    const float* x  = (const float*)d_in[0];
    const float* Wq = (const float*)d_in[1];
    const float* Wk = (const float*)d_in[2];
    const float* Wv = (const float*)d_in[3];
    const float* Wo = (const float*)d_in[4];
    float* out = (float*)d_out;

    const size_t headElems = (size_t)BATCH * NHEADS * SEQ * HDIM;   // 4M elems
    bf16_t* qws = (bf16_t*)d_ws;
    bf16_t* kws = qws + headElems;
    bf16_t* vws = kws + headElems;
    bf16_t* aws = vws + headElems;                                  // (B,S,D) bf16

    dim3 g1(DIM / 64, MTOT / 128, 3);
    qkv_rope_gemm<<<g1, 128, 0, stream>>>(x, Wq, Wk, Wv, qws, kws, vws);

    dim3 g2(SEQ / 64, NHEADS, BATCH);
    attention_kernel<<<g2, 128, 0, stream>>>(qws, kws, vws, aws);

    dim3 g3(DIM / 64, MTOT / 128, 1);
    oproj_gemm<<<g3, 128, 0, stream>>>(aws, Wo, out);
}
